// InterAgg_54511724921156
// MI455X (gfx1250) — matrix operand order repaired and checked by reading, not compiled, not run on hardware
//
#include <hip/hip_runtime.h>
#include <hip/hip_bf16.h>

typedef __attribute__((ext_vector_type(16))) __bf16 v16bf;
typedef __attribute__((ext_vector_type(8)))  __bf16 v8bf;
typedef __attribute__((ext_vector_type(4)))  __bf16 v4bf;
typedef __attribute__((ext_vector_type(8)))  float  v8f;
typedef __attribute__((ext_vector_type(4)))  float  v4f;

#define NN    65536   // nodes
#define KD    512     // feature dim
#define ED    512     // embed dim
#define G     32      // nodes per block
#define NBLK  (NN / G)  // 2048
#define YSTR  520     // bf16 elems per y row (pad: 520*2B=1040B, 260 dwords, 260%64==4)
#define HSTR2 261     // f32 elems per h row (261%64==5 -> conflict-free column reads)
#define SLOPE 0.2f

// ---------------- workspace layout ----------------
// [0,       524288)  Wbf      : 512*512 bf16
// [524288,  528384)  wa       : 1024 f32 (wa_c[512], wa_n[512])
// [528384,  552960)  partials : NBLK*3 f32 (per-block attention column sums)

__global__ __launch_bounds__(256)
void prep_kernel(const float* __restrict__ W, const float* __restrict__ a,
                 __bf16* __restrict__ Wbf, float* __restrict__ wa) {
    const int b = blockIdx.x, t = threadIdx.x;
    if (b < 1024) {
        const int idx = b * 256 + t;           // 262144 elems of W
        Wbf[idx] = (__bf16)W[idx];
    } else {
        const int j = (b - 1024) * 256 + t;    // 0..1023
        const int k = j & 511, which = j >> 9; // which: 0 = a_c, 1 = a_n
        const float* av = a + which * 512;
        float s = 0.f;
        for (int e = 0; e < 512; ++e) s += W[e * 512 + k] * av[e];
        wa[j] = s;                             // wa[k] = sum_e a[e] * W[e,k]
    }
}

__global__ __launch_bounds__(256)
void interagg_kernel(const float* __restrict__ self_f,
                     const float* __restrict__ neigh_f,
                     const __bf16* __restrict__ Wbf,
                     const float* __restrict__ wa,
                     float* __restrict__ out,
                     float* __restrict__ partials) {
    // s_buf aliases: bf16 y-panel [G][YSTR] (33280B) during GEMM,
    // then f32 h-halves [G][HSTR2] (33408B) during the transposed store.
    __shared__ __align__(16) char s_buf[G * HSTR2 * 4];
    __shared__ __align__(16) float s_wa[1024];
    __shared__ float s_e[4 * G];   // 128 row dots
    __shared__ float s_w[3 * G];   // attention weights

    __bf16* s_y = (__bf16*)s_buf;

    const int tid  = threadIdx.x;
    const int wv   = tid >> 5;        // wave 0..7
    const int lane = tid & 31;
    const int lr   = lane & 15;
    const int lh   = lane >> 4;
    const int n0   = blockIdx.x * G;

    // stage wa vectors
    for (int i = tid; i < 1024; i += 256) s_wa[i] = wa[i];
    __syncthreads();

    // ---- Phase 1: e-values. Row r: group g=r/32 (0=self,1..3=neigh rel), node i=r%32.
    // Wave w handles rows 16w..16w+15; lanes cooperate on one row (coalesced v4 loads).
    for (int rr = 0; rr < 16; ++rr) {
        const int row = wv * 16 + rr;
        const int g = row >> 5, i = row & 31;
        const float* src = (g == 0)
            ? (self_f  + (size_t)(n0 + i) * KD)
            : (neigh_f + ((size_t)(g - 1) * NN + n0 + i) * KD);
        const v4f* s4 = (const v4f*)src;
        const v4f* w4 = (const v4f*)(s_wa + (g == 0 ? 0 : 512));
        float acc = 0.f;
        #pragma unroll
        for (int j = 0; j < 4; ++j) {
            v4f x = s4[lane + 32 * j];
            v4f w = w4[lane + 32 * j];
            acc += x.x * w.x + x.y * w.y + x.z * w.z + x.w * w.w;
        }
        #pragma unroll
        for (int off = 16; off; off >>= 1) acc += __shfl_xor(acc, off, 32);
        if (lane == 0) s_e[row] = acc;
    }
    __syncthreads();

    // ---- Phase 2: per-node softmax over 3 relations (wave 0), block partial sums.
    if (tid < 32) {
        const int i = tid;
        const float ec = s_e[i];
        float v, l0, l1, l2;
        v = ec + s_e[32 + i]; l0 = v > 0.f ? v : SLOPE * v;
        v = ec + s_e[64 + i]; l1 = v > 0.f ? v : SLOPE * v;
        v = ec + s_e[96 + i]; l2 = v > 0.f ? v : SLOPE * v;
        const float m  = fmaxf(l0, fmaxf(l1, l2));
        const float p0 = __expf(l0 - m), p1 = __expf(l1 - m), p2 = __expf(l2 - m);
        const float inv = 1.f / (p0 + p1 + p2);
        const float w0 = p0 * inv, w1 = p1 * inv, w2 = p2 * inv;
        s_w[0 * G + i] = w0; s_w[1 * G + i] = w1; s_w[2 * G + i] = w2;
        float t0 = w0, t1 = w1, t2 = w2;
        #pragma unroll
        for (int off = 16; off; off >>= 1) {
            t0 += __shfl_xor(t0, off, 32);
            t1 += __shfl_xor(t1, off, 32);
            t2 += __shfl_xor(t2, off, 32);
        }
        if (i == 0) {
            partials[blockIdx.x * 3 + 0] = t0;
            partials[blockIdx.x * 3 + 1] = t1;
            partials[blockIdx.x * 3 + 2] = t2;
        }
    }
    __syncthreads();

    // ---- Phase 3: mix in feature space: y = x_self + sum_r w_r * x_neigh_r (bf16 LDS)
    for (int idx = tid; idx < G * 128; idx += 256) {     // float4 granules
        const int i = idx >> 7, k4 = idx & 127;
        const v4f* xs = (const v4f*)(self_f + (size_t)(n0 + i) * KD);
        const v4f* x0 = (const v4f*)(neigh_f + ((size_t)0 * NN + n0 + i) * KD);
        const v4f* x1 = (const v4f*)(neigh_f + ((size_t)1 * NN + n0 + i) * KD);
        const v4f* x2 = (const v4f*)(neigh_f + ((size_t)2 * NN + n0 + i) * KD);
        const float w0 = s_w[i], w1 = s_w[G + i], w2 = s_w[2 * G + i];
        v4f y = xs[k4];
        v4f a0 = x0[k4], a1 = x1[k4], a2 = x2[k4];
        y.x += w0 * a0.x + w1 * a1.x + w2 * a2.x;
        y.y += w0 * a0.y + w1 * a1.y + w2 * a2.y;
        y.z += w0 * a0.z + w1 * a1.z + w2 * a2.z;
        y.w += w0 * a0.w + w1 * a1.w + w2 * a2.w;
        v4bf yb;
        yb.x = (__bf16)y.x; yb.y = (__bf16)y.y; yb.z = (__bf16)y.z; yb.w = (__bf16)y.w;
        *(v4bf*)(s_y + (size_t)i * YSTR + 4 * k4) = yb;
    }
    __syncthreads();

    // ---- Phase 4: combined = y @ W^T via V_WMMA_F32_16X16X32_BF16.
    // 32 rows x 512 cols = 2 row-tiles x 32 col-tiles; wave w owns col-tiles {w, w+8, w+16, w+24}.
    union AF { v16bf v; v8bf h[2]; };
    v8f acc[2][4];
    const v8f vzero = {0.f, 0.f, 0.f, 0.f, 0.f, 0.f, 0.f, 0.f};
    #pragma unroll
    for (int mt = 0; mt < 2; ++mt)
        #pragma unroll
        for (int cc = 0; cc < 4; ++cc) acc[mt][cc] = vzero;

    for (int kc = 0; kc < 16; ++kc) {
        const int k0 = kc * 32;
        AF af[2];
        #pragma unroll
        for (int mt = 0; mt < 2; ++mt) {
            // A 16x32 layout (matches B): lane holds K run [base..base+15],
            // base = k0 + (lane/16)*16, row = M = lane%16.
            const __bf16* ap = s_y + (size_t)(mt * 16 + lr) * YSTR + k0 + lh * 16;
            af[mt].h[0] = *(const v8bf*)ap;
            af[mt].h[1] = *(const v8bf*)(ap + 8);
        }
        #pragma unroll
        for (int cc = 0; cc < 4; ++cc) {
            // B 32x16 layout (contiguous K, same as A): lane n holds K=(lane/16)*16 .. +15 of column n.
            // B[k,e] = W[e,k], W row-major -> 16 contiguous bf16 from Wbf row e.
            const int e = (wv + 8 * cc) * 16 + lr;
            const v16bf bfrag = *(const v16bf*)(Wbf + (size_t)e * KD + k0 + lh * 16);
            acc[0][cc] = __builtin_amdgcn_wmma_f32_16x16x32_bf16(
                false, af[0].v, false, bfrag, (short)0, acc[0][cc], false, false);
            acc[1][cc] = __builtin_amdgcn_wmma_f32_16x16x32_bf16(
                false, af[1].v, false, bfrag, (short)0, acc[1][cc], false, false);
        }
    }

    // ---- Phase 5: relu + transposed store via LDS (two e-halves; s_buf re-used as f32 h).
    float* s_h = (float*)s_buf;
    for (int half = 0; half < 2; ++half) {
        __syncthreads();   // half 0: all waves done reading s_y; half 1: done reading s_h
        #pragma unroll
        for (int mt = 0; mt < 2; ++mt)
            #pragma unroll
            for (int c2 = 0; c2 < 2; ++c2) {
                const int cc = half * 2 + c2;
                const int e0 = (wv + 8 * cc) * 16 - half * 256;
                const int rbase = mt * 16 + 8 * lh;   // C/D layout: VGPR i -> M=i (+8 hi half)
                #pragma unroll
                for (int i = 0; i < 8; ++i)
                    s_h[(size_t)(rbase + i) * HSTR2 + e0 + lr] = acc[mt][cc][i];
            }
        __syncthreads();
        // coalesced: each wave writes one 128B line (32 nodes) per e
        for (int er = 0; er < 256; er += 8) {
            const int e = er + wv;
            const float v = s_h[(size_t)lane * HSTR2 + e];
            out[(size_t)(half * 256 + e) * NN + n0 + lane] = fmaxf(v, 0.f);
        }
    }
}

__global__ __launch_bounds__(256)
void att_kernel(const float* __restrict__ partials, float* __restrict__ out) {
    __shared__ float sm[3][256];
    const int t = threadIdx.x;
    float s0 = 0.f, s1 = 0.f, s2 = 0.f;
    for (int b = t; b < NBLK; b += 256) {
        s0 += partials[b * 3 + 0];
        s1 += partials[b * 3 + 1];
        s2 += partials[b * 3 + 2];
    }
    sm[0][t] = s0; sm[1][t] = s1; sm[2][t] = s2;
    __syncthreads();
    for (int off = 128; off; off >>= 1) {
        if (t < off) {
            sm[0][t] += sm[0][t + off];
            sm[1][t] += sm[1][t + off];
            sm[2][t] += sm[2][t + off];
        }
        __syncthreads();
    }
    if (t == 0) {
        const float a0 = sm[0][0], a1 = sm[1][0], a2 = sm[2][0];
        const float m = fmaxf(a0, fmaxf(a1, a2));
        const float p0 = __expf(a0 - m), p1 = __expf(a1 - m), p2 = __expf(a2 - m);
        const float inv = 1.f / (p0 + p1 + p2);
        out[(size_t)ED * NN + 0] = p0 * inv;
        out[(size_t)ED * NN + 1] = p1 * inv;
        out[(size_t)ED * NN + 2] = p2 * inv;
    }
}

extern "C" void kernel_launch(void* const* d_in, const int* in_sizes, int n_in,
                              void* d_out, int out_size, void* d_ws, size_t ws_size,
                              hipStream_t stream) {
    const float* self_f  = (const float*)d_in[0];   // (N, 512)
    const float* neigh_f = (const float*)d_in[1];   // (3N, 512)
    const float* W       = (const float*)d_in[2];   // (512, 512)
    const float* a       = (const float*)d_in[3];   // (1024, 1)
    float* out = (float*)d_out;                     // 512*65536 combined + 3 att

    char* ws = (char*)d_ws;
    __bf16* Wbf     = (__bf16*)ws;                  // 524288 B
    float*  wa      = (float*)(ws + 524288);        // 4096 B
    float*  partial = (float*)(ws + 528384);        // NBLK*3*4 = 24576 B

    prep_kernel<<<1028, 256, 0, stream>>>(W, a, Wbf, wa);
    interagg_kernel<<<NBLK, 256, 0, stream>>>(self_f, neigh_f, Wbf, wa, out, partial);
    att_kernel<<<1, 256, 0, stream>>>(partial, out);
}
